// KernelVelocity_71201967833614
// MI455X (gfx1250) — compile-verified, matrix operand order checked by reading, not run on hardware
//
#include <hip/hip_runtime.h>

// Problem constants (match reference)
#define BATCH 512
#define NPTS  16384
#define DIM   2048
#define TOPM  64
#define KCH   32      // K per WMMA chunk (bf16 16x16x32)

typedef __attribute__((ext_vector_type(16))) __bf16          v16bf;
typedef __attribute__((ext_vector_type(16))) unsigned short  v16u;
typedef __attribute__((ext_vector_type(8)))  float           v8f;

// Swizzled ushort offset inside a 64x32 tile so that each lane's WMMA
// fragment (16 bf16) is 32 contiguous, 32B-aligned bytes:
//   g = row>>4, l15 = row&15, half = (k>>3)&1, j = (k&7) + ((k>>4)<<3)
//   off = ((g*32 + half*16 + l15) << 4) + j
__device__ __forceinline__ int soff(int row, int k) {
  return (((row >> 4) * 32 + ((k >> 3) & 1) * 16 + (row & 15)) << 4)
         + (k & 7) + ((k >> 4) << 3);
}

// split f32 -> bf16 hi/lo using the native hardware converts (RNE), then
// pack 4 halves into a uint2 so each tile store is one ds_store_b64.
__device__ __forceinline__ void put4s(unsigned short* __restrict__ H,
                                      unsigned short* __restrict__ L,
                                      int off, float4 v) {
  float vv[4] = {v.x, v.y, v.z, v.w};
  unsigned short hs[4], ls[4];
#pragma unroll
  for (int i = 0; i < 4; ++i) {
    const __bf16 h = (__bf16)vv[i];               // native cvt, round-nearest-even
    hs[i] = __builtin_bit_cast(unsigned short, h);
    const __bf16 l = (__bf16)(vv[i] - (float)h);  // residual term
    ls[i] = __builtin_bit_cast(unsigned short, l);
  }
  uint2 hp, lp;
  hp.x = (unsigned)hs[0] | ((unsigned)hs[1] << 16);
  hp.y = (unsigned)hs[2] | ((unsigned)hs[3] << 16);
  lp.x = (unsigned)ls[0] | ((unsigned)ls[1] << 16);
  lp.y = (unsigned)ls[2] | ((unsigned)ls[3] << 16);
  *(uint2*)(H + off) = hp;                        // off is 4-ushort aligned
  *(uint2*)(L + off) = lp;
}

__device__ __forceinline__ v8f wmma_bf16(v16bf a, v16bf b, v8f c) {
  return __builtin_amdgcn_wmma_f32_16x16x32_bf16(
      /*neg_a=*/false, a, /*neg_b=*/false, b,
      /*c_mod=*/(short)0, c, /*reuse_a=*/false, /*reuse_b=*/false);
}

// ---------------------------------------------------------------------------
// Kernel 1: fused x_t interpolation + bf16x3 WMMA GEMM + norms + exp kernel
//   kern[b,n] = exp(-max(|z_b|^2 + |x_t_n|^2 - 2 z_b.x_t_n, 0) / 2)
// grid: (NPTS/64, BATCH/64), block: 256 (8 waves, 4x2 wave grid, 16x32/wave)
// Double-buffered LDS, one barrier per K chunk; global loads of chunk k+1
// are issued before the WMMAs of chunk k (latency hiding).
// ---------------------------------------------------------------------------
__global__ __launch_bounds__(256)
void velo_gemm_kern(const float* __restrict__ z, const float* __restrict__ x0,
                    const float* __restrict__ x1, const float* __restrict__ tp,
                    float* __restrict__ kern) {
  // 8 swizzled 4KB tiles: {A,B} x {hi,lo} x double-buffer = 32 KB
  __shared__ v16u AhB[2][128], AlB[2][128], BhB[2][128], BlB[2][128];
  __shared__ float zn[64], xn[64];

  const int tid  = threadIdx.x;
  const int lane = tid & 31;
  const int wid  = tid >> 5;
  const int wm   = wid >> 1;          // 0..3 -> 16-row slice of A
  const int wn   = wid & 1;           // 0..1 -> 32-col slice of B
  const int b0   = blockIdx.y * 64;
  const int n0   = blockIdx.x * 64;

  const float t  = tp[0];
  const float c0 = 1.0f - t, c1 = t;

  const int fr   = tid >> 3;          // staging rows fr and fr+32
  const int fc   = (tid & 7) << 2;    // k offset (float4 granular)
  const int off0 = soff(fr, fc);
  const int off1 = soff(fr + 32, fc);

  if (tid < 64) { zn[tid] = 0.0f; xn[tid] = 0.0f; }

  v8f acc0 = {}; v8f acc1 = {};
  float znp0 = 0.f, znp1 = 0.f, xnp0 = 0.f, xnp1 = 0.f;

  // ---- prologue: load + stage chunk 0 into buffer 0 ----
  float4 za, zb, xu, xv;
  {
    const int k0 = 0;
    za = *(const float4*)(z + (size_t)(b0 + fr)      * DIM + k0 + fc);
    zb = *(const float4*)(z + (size_t)(b0 + fr + 32) * DIM + k0 + fc);
    const size_t r0 = (size_t)(n0 + fr) * DIM + k0 + fc;
    const size_t r1 = (size_t)(n0 + fr + 32) * DIM + k0 + fc;
    const float4 p0 = *(const float4*)(x0 + r0);
    const float4 q0 = *(const float4*)(x1 + r0);
    const float4 p1 = *(const float4*)(x0 + r1);
    const float4 q1 = *(const float4*)(x1 + r1);
    xu.x = c0*p0.x + c1*q0.x; xu.y = c0*p0.y + c1*q0.y;
    xu.z = c0*p0.z + c1*q0.z; xu.w = c0*p0.w + c1*q0.w;
    xv.x = c0*p1.x + c1*q1.x; xv.y = c0*p1.y + c1*q1.y;
    xv.z = c0*p1.z + c1*q1.z; xv.w = c0*p1.w + c1*q1.w;
    znp0 += za.x*za.x + za.y*za.y + za.z*za.z + za.w*za.w;
    znp1 += zb.x*zb.x + zb.y*zb.y + zb.z*zb.z + zb.w*zb.w;
    xnp0 += xu.x*xu.x + xu.y*xu.y + xu.z*xu.z + xu.w*xu.w;
    xnp1 += xv.x*xv.x + xv.y*xv.y + xv.z*xv.z + xv.w*xv.w;
    put4s((unsigned short*)&AhB[0][0], (unsigned short*)&AlB[0][0], off0, za);
    put4s((unsigned short*)&AhB[0][0], (unsigned short*)&AlB[0][0], off1, zb);
    put4s((unsigned short*)&BhB[0][0], (unsigned short*)&BlB[0][0], off0, xu);
    put4s((unsigned short*)&BhB[0][0], (unsigned short*)&BlB[0][0], off1, xv);
  }

  for (int k0 = 0; k0 < DIM; k0 += KCH) {
    const int p = (k0 >> 5) & 1;
    __syncthreads();                               // buf[p] staged & readable

    // ---- issue global loads for the NEXT chunk first (latency hiding) ----
    const bool have = (k0 + KCH) < DIM;
    if (have) {
      const int kn = k0 + KCH;
      za = *(const float4*)(z + (size_t)(b0 + fr)      * DIM + kn + fc);
      zb = *(const float4*)(z + (size_t)(b0 + fr + 32) * DIM + kn + fc);
      const size_t r0 = (size_t)(n0 + fr) * DIM + kn + fc;
      const size_t r1 = (size_t)(n0 + fr + 32) * DIM + kn + fc;
      const float4 p0 = *(const float4*)(x0 + r0);
      const float4 q0 = *(const float4*)(x1 + r0);
      const float4 p1 = *(const float4*)(x0 + r1);
      const float4 q1 = *(const float4*)(x1 + r1);
      xu.x = c0*p0.x + c1*q0.x; xu.y = c0*p0.y + c1*q0.y;
      xu.z = c0*p0.z + c1*q0.z; xu.w = c0*p0.w + c1*q0.w;
      xv.x = c0*p1.x + c1*q1.x; xv.y = c0*p1.y + c1*q1.y;
      xv.z = c0*p1.z + c1*q1.z; xv.w = c0*p1.w + c1*q1.w;
    }

    // ---- fragments: one 32B contiguous LDS vector read per operand ----
    const v16bf a_h = __builtin_bit_cast(v16bf, AhB[p][wm * 32 + lane]);
    const v16bf a_l = __builtin_bit_cast(v16bf, AlB[p][wm * 32 + lane]);
    const v16bf b0h = __builtin_bit_cast(v16bf, BhB[p][wn * 64 + lane]);
    const v16bf b0l = __builtin_bit_cast(v16bf, BlB[p][wn * 64 + lane]);
    const v16bf b1h = __builtin_bit_cast(v16bf, BhB[p][wn * 64 + 32 + lane]);
    const v16bf b1l = __builtin_bit_cast(v16bf, BlB[p][wn * 64 + 32 + lane]);

    // bf16x3: z.x ~= zh.xh + zh.xl + zl.xh  (near-fp32 ranking fidelity)
    acc0 = wmma_bf16(a_h, b0h, acc0);
    acc0 = wmma_bf16(a_h, b0l, acc0);
    acc0 = wmma_bf16(a_l, b0h, acc0);
    acc1 = wmma_bf16(a_h, b1h, acc1);
    acc1 = wmma_bf16(a_h, b1l, acc1);
    acc1 = wmma_bf16(a_l, b1h, acc1);

    // ---- convert + stage next chunk into the other buffer ----
    if (have) {
      znp0 += za.x*za.x + za.y*za.y + za.z*za.z + za.w*za.w;
      znp1 += zb.x*zb.x + zb.y*zb.y + zb.z*zb.z + zb.w*zb.w;
      xnp0 += xu.x*xu.x + xu.y*xu.y + xu.z*xu.z + xu.w*xu.w;
      xnp1 += xv.x*xv.x + xv.y*xv.y + xv.z*xv.z + xv.w*xv.w;
      const int q = p ^ 1;
      put4s((unsigned short*)&AhB[q][0], (unsigned short*)&AlB[q][0], off0, za);
      put4s((unsigned short*)&AhB[q][0], (unsigned short*)&AlB[q][0], off1, zb);
      put4s((unsigned short*)&BhB[q][0], (unsigned short*)&BlB[q][0], off0, xu);
      put4s((unsigned short*)&BhB[q][0], (unsigned short*)&BlB[q][0], off1, xv);
    }
  }

  __syncthreads();
  atomicAdd(&zn[fr],      znp0);
  atomicAdd(&zn[fr + 32], znp1);
  atomicAdd(&xn[fr],      xnp0);
  atomicAdd(&xn[fr + 32], xnp1);
  __syncthreads();

  // ---- epilogue: sq -> Gaussian kernel (H = 1 -> /2) ----
  // C layout: VGPR r holds M = r + 8*(lane>>4), N = lane&15
  const int half = lane >> 4;
  const int l15  = lane & 15;
#pragma unroll
  for (int r = 0; r < 8; ++r) {
    const int rl = wm * 16 + r + half * 8;
    {
      const int cl = wn * 32 + l15;
      const float sq = zn[rl] + xn[cl] - 2.0f * acc0[r];
      kern[(size_t)(b0 + rl) * NPTS + (n0 + cl)] = __expf(-fmaxf(sq, 0.0f) * 0.5f);
    }
    {
      const int cl = wn * 32 + 16 + l15;
      const float sq = zn[rl] + xn[cl] - 2.0f * acc1[r];
      kern[(size_t)(b0 + rl) * NPTS + (n0 + cl)] = __expf(-fmaxf(sq, 0.0f) * 0.5f);
    }
  }
}

// ---------------------------------------------------------------------------
// Kernel 2: per-batch-row top-64 (lowest-index tie-break, matching
// jax.lax.top_k) + normalized-weight gather of x_1 -> velocity
// grid: BATCH, block: 256, dynamic LDS: NPTS floats (64 KB of 320 KB WGP LDS)
// ---------------------------------------------------------------------------
__global__ __launch_bounds__(256)
void velo_topk_kern(const float* __restrict__ kern, const float* __restrict__ z,
                    const float* __restrict__ x1, const float* __restrict__ tp,
                    float* __restrict__ out) {
  extern __shared__ float vals[];                 // NPTS
  __shared__ float redv[256];
  __shared__ int   redi[256];
  __shared__ float topv[TOPM];
  __shared__ int   topi[TOPM];
  __shared__ float ssum;

  const int tid = threadIdx.x;
  const int b   = blockIdx.x;

  const float* row = kern + (size_t)b * NPTS;
  for (int i = tid; i < NPTS; i += 256) vals[i] = row[i];
  __syncthreads();

  for (int m = 0; m < TOPM; ++m) {
    float bv = -1.0f; int bi = NPTS;              // kern >= 0, so -1 == -inf
    for (int j = 0; j < NPTS / 256; ++j) {
      const int idx = tid + (j << 8);
      const float v = vals[idx];
      if (v > bv || (v == bv && idx < bi)) { bv = v; bi = idx; }
    }
    redv[tid] = bv; redi[tid] = bi;
    __syncthreads();
    for (int s = 128; s > 0; s >>= 1) {
      if (tid < s) {
        const float ov = redv[tid + s]; const int oi = redi[tid + s];
        if (ov > redv[tid] || (ov == redv[tid] && oi < redi[tid])) {
          redv[tid] = ov; redi[tid] = oi;
        }
      }
      __syncthreads();
    }
    if (tid == 0) {
      topv[m] = redv[0]; topi[m] = redi[0];
      vals[redi[0]] = -1.0f;                      // knock out the winner
    }
    __syncthreads();
  }

  if (tid == 0) {
    float s = 0.0f;
    for (int m = 0; m < TOPM; ++m) s += topv[m];
    ssum = s;
  }
  __syncthreads();

  const float t    = tp[0];
  const float invd = 1.0f / (ssum + 1e-7f);       // w_m = topv_m * invd
  const float wsum = ssum * invd;                 // sum of normalized weights
  const float sc   = 1.0f / (1.0f - t + 1e-7f);

  float acc[DIM / 256];
#pragma unroll
  for (int k = 0; k < DIM / 256; ++k) acc[k] = 0.0f;

  for (int m = 0; m < TOPM; ++m) {
    const float wgt = topv[m] * invd;
    const float* xr = x1 + (size_t)topi[m] * DIM;
    if (m + 1 < TOPM)                             // global_prefetch_b8 next row
      __builtin_prefetch(x1 + (size_t)topi[m + 1] * DIM + tid, 0, 1);
#pragma unroll
    for (int k = 0; k < DIM / 256; ++k)
      acc[k] += wgt * xr[tid + (k << 8)];         // coalesced across lanes
  }

  const float* zr = z + (size_t)b * DIM;
#pragma unroll
  for (int k = 0; k < DIM / 256; ++k) {
    const int d = tid + (k << 8);
    out[(size_t)b * DIM + d] = (acc[k] - zr[d] * wsum) * sc;
  }
}

// ---------------------------------------------------------------------------
extern "C" void kernel_launch(void* const* d_in, const int* in_sizes, int n_in,
                              void* d_out, int out_size, void* d_ws, size_t ws_size,
                              hipStream_t stream) {
  const float* z  = (const float*)d_in[0];   // [512, 2048]
  const float* x0 = (const float*)d_in[1];   // [16384, 2048]
  const float* x1 = (const float*)d_in[2];   // [16384, 2048]
  const float* tp = (const float*)d_in[3];   // scalar t
  float* out  = (float*)d_out;               // [512, 2048]
  float* kern = (float*)d_ws;                // [512, 16384] scratch (32 MB)

  dim3 g1(NPTS / 64, BATCH / 64);            // (256, 8)
  velo_gemm_kern<<<g1, 256, 0, stream>>>(z, x0, x1, tp, kern);

  velo_topk_kern<<<BATCH, 256, NPTS * sizeof(float), stream>>>(kern, z, x1, tp, out);
}
